// Graphormer_AttentionHead_27839978013467
// MI455X (gfx1250) — matrix operand order repaired and checked by reading, not compiled, not run on hardware
//
#include <hip/hip_runtime.h>
#include <hip/hip_bf16.h>

// ---------------------------------------------------------------------------
// Graphormer attention head, MI455X (gfx1250, wave32, WMMA bf16 path)
//   N=8192, HIDDEN=768, DK=64
// Memory-bound by the 256MB fp32 bias matrix -> flash-attention structure,
// bf16 WMMA for all GEMMs, fp32 (base-2) online softmax.
// This revision computes S^T (A=K, B=Q) so softmax rows live along the
// register axis: per-lane reduction trees + one xor-16 shuffle, scalar m/l.
// ---------------------------------------------------------------------------

constexpr int N   = 8192;
constexpr int HID = 768;
constexpr int DK  = 64;

#define LOG2E 1.44269504088896340736f

typedef __attribute__((ext_vector_type(16))) __bf16 v16bf;
typedef __attribute__((ext_vector_type(2)))  __bf16 v2bf;
typedef __attribute__((ext_vector_type(8)))  float  v8f;

union BFrag { v16bf v; unsigned int u[8]; };

__device__ __forceinline__ unsigned short f32_to_bf16_rne(float f) {
    // native bf16 on gfx1250: fptrunc should lower to v_cvt*bf16*
    return __builtin_bit_cast(unsigned short, (__bf16)f);
}

__device__ __forceinline__ unsigned int pack_bf16(float lo, float hi) {
#if defined(__has_builtin) && __has_builtin(__builtin_amdgcn_cvt_pk_bf16_f32)
    v2bf t = __builtin_amdgcn_cvt_pk_bf16_f32(lo, hi);
    return __builtin_bit_cast(unsigned int, t);
#else
    v2bf t; t.x = (__bf16)lo; t.y = (__bf16)hi;
    return __builtin_bit_cast(unsigned int, t);
#endif
}

__device__ __forceinline__ float fast_exp2(float x) {
#if defined(__has_builtin) && __has_builtin(__builtin_amdgcn_exp2f)
    return __builtin_amdgcn_exp2f(x);
#else
    return exp2f(x);
#endif
}

__device__ __forceinline__ v8f wmma_bf16(const v16bf& a, const v16bf& b, const v8f& c) {
    // (neg_a, A, neg_b, B, c_mod, C, reuse_a, reuse_b)
    return __builtin_amdgcn_wmma_f32_16x16x32_bf16(false, a, false, b,
                                                   (short)0, c, false, false);
}

// ---------------------------------------------------------------------------
// Kernel 0: convert W -> bf16, transposed to Wt[mat][c][k] so that WMMA
// B-fragment pairs (along K) are contiguous in memory (b128-mergeable loads).
// ---------------------------------------------------------------------------
__global__ __launch_bounds__(256) void prep_w_kernel(
    const float* __restrict__ Wq, const float* __restrict__ Wk,
    const float* __restrict__ Wv, unsigned short* __restrict__ Wt)
{
    const int mat = blockIdx.y;
    const float* W = (mat == 0) ? Wq : (mat == 1) ? Wk : Wv;
    const int idx = blockIdx.x * 256 + threadIdx.x;   // over HID*DK
    const int k = idx / DK, c = idx % DK;             // W[k][c]
    Wt[(size_t)mat * DK * HID + (size_t)c * HID + k] = f32_to_bf16_rne(W[idx]);
}

// ---------------------------------------------------------------------------
// Kernel 1: Q/K/V projection.  Each wave: one 16-row tile x all 64 cols of one
// of Q/K/V (blockIdx.y selects which).  bf16 WMMA, fp32 accumulate.
//   Q row-major bf16, pre-scaled by log2(e)/sqrt(DK)  (used as B-fragments)
//   K row-major bf16                                   (used as A-fragments)
//   V transposed (Vt[d][key]) with packed dword stores (used as A-fragments)
// ---------------------------------------------------------------------------
__global__ __launch_bounds__(256) void qkv_kernel(
    const float* __restrict__ H, const unsigned short* __restrict__ Wt,
    const float* __restrict__ bq, const float* __restrict__ bk,
    const float* __restrict__ bv,
    unsigned short* __restrict__ Qb, unsigned short* __restrict__ Kb,
    unsigned short* __restrict__ Vt)
{
    const int mat = blockIdx.y;
    const float* bias = (mat == 0) ? bq : (mat == 1) ? bk : bv;
    const unsigned int* Wm =
        (const unsigned int*)(Wt + (size_t)mat * DK * HID);   // Wt[c][k] pairs
    const float oscale = (mat == 0) ? (0.125f * LOG2E) : 1.0f;

    const int wave = threadIdx.x >> 5;
    const int lane = threadIdx.x & 31;
    const int n15  = lane & 15;
    const int half = lane >> 4;
    const int row0 = (blockIdx.x * 8 + wave) * 16;

    v8f acc[4];
    #pragma unroll
    for (int nt = 0; nt < 4; ++nt) acc[nt] = (v8f){};

    for (int kk = 0; kk < HID; kk += 32) {
        // A fragment: H[row0+n15][kk + kb], kb = 2(j&3) + 16(j>>2) + 8*half
        BFrag a;
        const float* hrow = H + (size_t)(row0 + n15) * HID + kk + 8 * half;
        #pragma unroll
        for (int j = 0; j < 8; ++j) {
            const int kb = 2 * (j & 3) + 16 * (j >> 2);
            const float2 hv = *(const float2*)(hrow + kb);
            a.u[j] = pack_bf16(hv.x, hv.y);
        }
        // B fragments: Wt[c][kk + kb .. +1], kb = 2(j&3) + 16(j>>2) + 8*half
        #pragma unroll
        for (int nt = 0; nt < 4; ++nt) {
            BFrag b;
            const unsigned int* bp =
                Wm + (((size_t)(nt * 16 + n15) * HID + kk) >> 1) + 4 * half;
            #pragma unroll
            for (int j = 0; j < 8; ++j) b.u[j] = bp[(j & 3) + 8 * (j >> 2)];
            acc[nt] = wmma_bf16(a.v, b.v, acc[nt]);
        }
    }

    if (mat < 2) {
        unsigned short* out = (mat == 0) ? Qb : Kb;
        #pragma unroll
        for (int nt = 0; nt < 4; ++nt) {
            const float badd = bias[nt * 16 + n15];
            #pragma unroll
            for (int r = 0; r < 8; ++r) {
                const float v = (acc[nt][r] + badd) * oscale;
                out[(size_t)(row0 + r + 8 * half) * DK + nt * 16 + n15] =
                    f32_to_bf16_rne(v);
            }
        }
    } else {
        // Vt[c][row]: lane owns column c = nt*16+n15; rows row0+8*half+0..7
        // are contiguous -> 4 packed dword stores per nt.
        #pragma unroll
        for (int nt = 0; nt < 4; ++nt) {
            const float badd = bias[nt * 16 + n15];
            unsigned int* vp = (unsigned int*)Vt +
                (((size_t)(nt * 16 + n15) * N + row0) >> 1) + 4 * half;
            #pragma unroll
            for (int r2 = 0; r2 < 4; ++r2)
                vp[r2] = pack_bf16(acc[nt][2 * r2] + badd,
                                   acc[nt][2 * r2 + 1] + badd);
        }
    }
}

// ---------------------------------------------------------------------------
// Kernel 2: flash attention, transposed tiles.  One wave per 16 query rows,
// key tiles of 64.
//   S^T[key][q] : A = K tile (16 keys x 32 d, row-major, b128-mergeable),
//                 B = Q tile (32 d x 16 q, loop-invariant contiguous dwords)
//   C layout puts query on the lane axis, key on the register axis ->
//   bias/mask are per-lane-row float4/uint2 loads; softmax row reduction is a
//   per-lane register tree + one xor-16 shuffle; m/l/alpha are lane scalars.
//   P^T staged in per-wave LDS (packed dwords, stride-33 padding), reloaded
//   as contiguous B-fragments.  O^T += V^T x P^T with V^T straight from Vt.
// ---------------------------------------------------------------------------
__global__ __launch_bounds__(256) void attn_kernel(
    const unsigned short* __restrict__ Qb,
    const unsigned short* __restrict__ Kb,
    const unsigned short* __restrict__ Vt,
    const float* __restrict__ bmat,
    const unsigned char* __restrict__ mask,
    float* __restrict__ out)
{
    __shared__ unsigned int pShm[8][16 * 33];   // per-wave P^T staging

    const int wave = threadIdx.x >> 5;
    const int lane = threadIdx.x & 31;
    const int n15  = lane & 15;
    const int half = lane >> 4;
    const int q0   = (blockIdx.x * 8 + wave) * 16;

    // Q as B-fragments (32 d x 16 q): pairs along d, col = query. Invariant.
    BFrag qf[2];
    {
        const unsigned int* qrow =
            (const unsigned int*)Qb + (size_t)(q0 + n15) * 32;   // 32 u32/row
        #pragma unroll
        for (int dh = 0; dh < 2; ++dh)
            #pragma unroll
            for (int j = 0; j < 8; ++j)
                qf[dh].u[j] = qrow[dh * 16 + 4 * half + (j & 3) + 8 * (j >> 2)];
    }

    v8f o[4];
    #pragma unroll
    for (int nt = 0; nt < 4; ++nt) o[nt] = (v8f){};
    float mstate = -1.0e30f, lstate = 0.0f;

    unsigned int* myShm = pShm[wave];
    const float*         brow = bmat + (size_t)(q0 + n15) * N;
    const unsigned char* mrow = mask + (size_t)(q0 + n15) * N;

    for (int k0 = 0; k0 < N; k0 += 64) {
        // ---- S^T: 4 key-quadrant fragments, reg r -> key k0+16qq+8*half+r
        v8f s[4];
        #pragma unroll
        for (int qq = 0; qq < 4; ++qq) s[qq] = (v8f){};
        #pragma unroll
        for (int qq = 0; qq < 4; ++qq)
            #pragma unroll
            for (int dh = 0; dh < 2; ++dh) {
                BFrag a;
                const unsigned int* kp = (const unsigned int*)Kb +
                    (size_t)(k0 + 16 * qq + n15) * 32 + dh * 16 + 4 * half;
                #pragma unroll
                for (int j = 0; j < 8; ++j)
                    a.u[j] = kp[(j & 3) + 8 * (j >> 2)];
                s[qq] = wmma_bf16(a.v, qf[dh].v, s[qq]);
            }

        // ---- bias + mask: each lane streams its own query row (vectorized)
        #pragma unroll
        for (int qq = 0; qq < 4; ++qq) {
            const size_t off = (size_t)k0 + 16 * qq + 8 * half;
            const float4 b0 = *(const float4*)(brow + off);
            const float4 b1 = *(const float4*)(brow + off + 4);
            const uint2  mu = *(const uint2*)(mrow + off);
            const float bb[8] = {b0.x, b0.y, b0.z, b0.w, b1.x, b1.y, b1.z, b1.w};
            #pragma unroll
            for (int r = 0; r < 8; ++r) {
                const unsigned int mb =
                    ((r < 4 ? mu.x : mu.y) >> (8 * (r & 3))) & 0xffu;
                const float t = fmaf(bb[r], LOG2E, s[qq][r]);
                s[qq][r] = mb ? t : -1.0e30f;
            }
        }

        // ---- base-2 online softmax: per-lane register tree + one shuffle
        float mx = -1.0e30f;
        #pragma unroll
        for (int qq = 0; qq < 4; ++qq)
            #pragma unroll
            for (int r = 0; r < 8; ++r) mx = fmaxf(mx, s[qq][r]);
        mx = fmaxf(mx, __shfl_xor(mx, 16, 32));
        const float mnew  = fmaxf(mstate, mx);
        const float alpha = fast_exp2(mstate - mnew);
        float rs = 0.0f;
        #pragma unroll
        for (int qq = 0; qq < 4; ++qq)
            #pragma unroll
            for (int r = 0; r < 8; ++r) {
                const float p = fast_exp2(s[qq][r] - mnew);
                s[qq][r] = p;
                rs += p;
            }
        rs += __shfl_xor(rs, 16, 32);
        lstate = lstate * alpha + rs;
        mstate = mnew;
        #pragma unroll
        for (int nt = 0; nt < 4; ++nt)
            #pragma unroll
            for (int r = 0; r < 8; ++r) o[nt][r] *= alpha;

        // ---- P^T -> LDS [query][key] bf16 pairs (stride 33 dwords)
        #pragma unroll
        for (int qq = 0; qq < 4; ++qq)
            #pragma unroll
            for (int r2 = 0; r2 < 4; ++r2)
                myShm[n15 * 33 + 8 * qq + 4 * half + r2] =
                    pack_bf16(s[qq][2 * r2], s[qq][2 * r2 + 1]);
        __syncthreads();   // conservative fence (DS in-order within a wave)
        BFrag pf[2];
        #pragma unroll
        for (int c = 0; c < 2; ++c) {
            const unsigned int* ps = myShm + n15 * 33 + 16 * c + 4 * half;
            #pragma unroll
            for (int j = 0; j < 8; ++j) pf[c].u[j] = ps[(j & 3) + 8 * (j >> 2)];
        }
        __syncthreads();

        // ---- O^T += V^T x P^T : V^T A-fragments straight from Vt (b128)
        #pragma unroll
        for (int c = 0; c < 2; ++c)
            #pragma unroll
            for (int nt = 0; nt < 4; ++nt) {
                BFrag a;
                const unsigned int* vp = (const unsigned int*)Vt +
                    (((size_t)(nt * 16 + n15) * N + k0) >> 1) + 16 * c + 4 * half;
                #pragma unroll
                for (int j = 0; j < 8; ++j)
                    a.u[j] = vp[(j & 3) + 8 * (j >> 2)];
                o[nt] = wmma_bf16(a.v, pf[c].v, o[nt]);
            }
    }

    // ---- epilogue: per-lane scalar 1/l, vectorized float4 row stores
    const float inv = 1.0f / lstate;
    float* orow = out + (size_t)(q0 + n15) * DK + 8 * half;
    #pragma unroll
    for (int nt = 0; nt < 4; ++nt) {
        float4 v0 = {o[nt][0] * inv, o[nt][1] * inv, o[nt][2] * inv, o[nt][3] * inv};
        float4 v1 = {o[nt][4] * inv, o[nt][5] * inv, o[nt][6] * inv, o[nt][7] * inv};
        *(float4*)(orow + nt * 16)     = v0;
        *(float4*)(orow + nt * 16 + 4) = v1;
    }
}

// ---------------------------------------------------------------------------
extern "C" void kernel_launch(void* const* d_in, const int* in_sizes, int n_in,
                              void* d_out, int out_size, void* d_ws, size_t ws_size,
                              hipStream_t stream) {
    const float*         H    = (const float*)d_in[0];
    const float*         bmat = (const float*)d_in[1];
    const unsigned char* msk  = (const unsigned char*)d_in[2];
    const float*         Wq   = (const float*)d_in[3];
    const float*         bq   = (const float*)d_in[4];
    const float*         Wk   = (const float*)d_in[5];
    const float*         bk   = (const float*)d_in[6];
    const float*         Wv   = (const float*)d_in[7];
    const float*         bv   = (const float*)d_in[8];

    unsigned short* Qb = (unsigned short*)d_ws;           // N*DK bf16 (1MB)
    unsigned short* Kb = Qb + (size_t)N * DK;             // N*DK bf16 (1MB)
    unsigned short* Vt = Kb + (size_t)N * DK;             // DK*N bf16 (1MB)
    unsigned short* Wt = Vt + (size_t)N * DK;             // 3*DK*HID bf16

    prep_w_kernel<<<dim3(HID * DK / 256, 3), 256, 0, stream>>>(Wq, Wk, Wv, Wt);
    qkv_kernel<<<dim3(N / 128, 3), 256, 0, stream>>>(H, Wt, bq, bk, bv,
                                                     Qb, Kb, Vt);
    attn_kernel<<<dim3(N / 128), 256, 0, stream>>>(Qb, Kb, Vt, bmat, msk,
                                                   (float*)d_out);
}
